// ModifiedTransformerEncoderLayer_15212774163014
// MI455X (gfx1250) — matrix operand order repaired and checked by reading, not compiled, not run on hardware
//
#include <hip/hip_runtime.h>
#include <hip/hip_bf16.h>
#include <math.h>

typedef __attribute__((ext_vector_type(16))) _Float16 v16h;
typedef __attribute__((ext_vector_type(8)))  _Float16 v8h;
typedef __attribute__((ext_vector_type(8)))  float    v8f;

static constexpr int BB  = 4096;
static constexpr int JJ  = 17;
static constexpr int DD  = 512;
static constexpr int DK  = 64;
static constexpr int HH  = 8;
static constexpr int MM  = BB * JJ;      // 69632 (multiple of 128)

// ---------------------------------------------------------------------------
// Fragment-set loader: A fragment (16x32) + 8 B fragments (32x16 each).
// ---------------------------------------------------------------------------
__device__ __forceinline__
void load_set(const _Float16* __restrict__ Ap, const _Float16* __restrict__ Bp0,
              int ldb, int kk, v16h& af, v16h* bf)
{
    v8h a0 = *(const v8h*)(Ap + kk);
    v8h a1 = *(const v8h*)(Ap + kk + 16);
    #pragma unroll
    for (int i = 0; i < 8; ++i) { af[i] = a0[i]; af[i + 8] = a1[i]; }

    #pragma unroll
    for (int c = 0; c < 8; ++c) {
        const _Float16* Bp = Bp0 + (size_t)(c * 16) * ldb + kk;
        v8h b0 = *(const v8h*)(Bp);
        v8h b1 = *(const v8h*)(Bp + 16);
        #pragma unroll
        for (int i = 0; i < 8; ++i) { bf[c][i] = b0[i]; bf[c][i + 8] = b1[i]; }
    }
}

__device__ __forceinline__
void wmma_burst(const v16h& af, const v16h* bf, v8f* acc)
{
    #pragma unroll
    for (int c = 0; c < 8; ++c)
        acc[c] = __builtin_amdgcn_wmma_f32_16x16x32_f16(
            false, af, false, bf[c], (short)0, acc[c], false, false);
}

// ---------------------------------------------------------------------------
// Generic f16 WMMA GEMM:  C[M x N] = A[M x K](f16, row-major, lda)
//                                   x Bt[N x K](f16, B transposed, ldb=K)
//                                   + bias[N]   (f32, may be null)
// Block = 256 threads (8 waves). Block tile = 128x128, wave tile = 16x128.
// Software-pipelined: loads for k-step i+1 issued before WMMAs of k-step i,
// forcing two live fragment sets so loads overlap the WMMA burst.
// K must be a multiple of 64 (512 / 512 / 1024 here).
// ---------------------------------------------------------------------------
template<int OUT_HALF>
__global__ __launch_bounds__(256)
void gemm_wmma_f16(const _Float16* __restrict__ A, int lda,
                   const _Float16* __restrict__ Bt, int ldb,
                   const float* __restrict__ bias,
                   void* __restrict__ Cv, int ldc, int K)
{
    const int lane = threadIdx.x & 31;
    const int wave = threadIdx.x >> 5;
    const int m0   = blockIdx.x * 128 + wave * 16;
    const int n0   = blockIdx.y * 128;
    const int hi   = lane >> 4;        // 0: lanes 0-15, 1: lanes 16-31
    const int l15  = lane & 15;

    v8f acc[8];
    #pragma unroll
    for (int c = 0; c < 8; ++c)
        #pragma unroll
        for (int i = 0; i < 8; ++i) acc[c][i] = 0.0f;

    // A row private to this lane; lanes >=16 start 8 halves in (K layout)
    const _Float16* Ap = A + (size_t)(m0 + l15) * lda + hi * 8;
    // B fragment base: lane = column, K contiguous (transposed weights)
    const _Float16* Bp0 = Bt + (size_t)(n0 + l15) * ldb + hi * 8;

    v16h afA, afB;
    v16h bfA[8], bfB[8];

    load_set(Ap, Bp0, ldb, 0, afA, bfA);

    int kk = 0;
    for (; kk + 64 < K; kk += 64) {
        __builtin_prefetch(Ap + kk + 128, 0, 1);       // global_prefetch_b8
        load_set(Ap, Bp0, ldb, kk + 32, afB, bfB);
        wmma_burst(afA, bfA, acc);
        load_set(Ap, Bp0, ldb, kk + 64, afA, bfA);
        wmma_burst(afB, bfB, acc);
    }
    load_set(Ap, Bp0, ldb, K - 32, afB, bfB);
    wmma_burst(afA, bfA, acc);
    wmma_burst(afB, bfB, acc);

    // Epilogue: VGPR r -> row m0 + r + (hi?8:0), col n0 + c*16 + l15
    const int rbase = m0 + hi * 8;
    #pragma unroll
    for (int c = 0; c < 8; ++c) {
        const int col = n0 + c * 16 + l15;
        const float bi = bias ? bias[col] : 0.0f;
        #pragma unroll
        for (int r = 0; r < 8; ++r) {
            float v = acc[c][r] + bi;
            size_t off = (size_t)(rbase + r) * ldc + col;
            if (OUT_HALF) ((_Float16*)Cv)[off] = (_Float16)v;
            else          ((float*)Cv)[off]    = v;
        }
    }
}

// ---------------------------------------------------------------------------
// Weight / input prep kernels (fp32 -> f16, transposed for the GEMM B side)
// ---------------------------------------------------------------------------
__global__ void k_cvt_src(const float* __restrict__ x, _Float16* __restrict__ y, int n)
{
    int i = blockIdx.x * 256 + threadIdx.x;
    if (i < n) y[i] = (_Float16)x[i];
}

// WqkvT[n][d], n = s*512 + h*64 + e  (s: 0=Q,1=K,2=V), d = 0..511
__global__ void k_build_wqkvT(const float* __restrict__ Wq, const float* __restrict__ Wk,
                              const float* __restrict__ Wv, _Float16* __restrict__ out)
{
    int idx = blockIdx.x * 256 + threadIdx.x;          // 1536*512
    if (idx >= 1536 * 512) return;
    int n = idx >> 9, d = idx & 511;
    int s = n >> 9, hc = n & 511, h = hc >> 6, e = hc & 63;
    const float* W = (s == 0) ? Wq : (s == 1) ? Wk : Wv;   // [H,D,DK]
    out[(size_t)n * 512 + d] = (_Float16)W[((size_t)h * 512 + d) * 64 + e];
}

__global__ void k_build_bqkv(const float* __restrict__ bq, const float* __restrict__ bk,
                             const float* __restrict__ bv, float* __restrict__ out)
{
    int n = blockIdx.x * 256 + threadIdx.x;
    if (n >= 1536) return;
    int s = n >> 9, hc = n & 511;
    const float* b = (s == 0) ? bq : (s == 1) ? bk : bv;   // [H,DK] flat 512
    out[n] = b[hc];
}

// WoT[n][k] = Wo[k][n], Wo row-major [512 x 512]
__global__ void k_build_woT(const float* __restrict__ Wo, _Float16* __restrict__ out)
{
    int idx = blockIdx.x * 256 + threadIdx.x;          // 512*512
    if (idx >= 512 * 512) return;
    int n = idx >> 9, k = idx & 511;
    out[(size_t)n * 512 + k] = (_Float16)Wo[(size_t)k * 512 + n];
}

// W01T[n][k], k<512 -> W_gcn[0][k][n], k>=512 -> W_gcn[1][k-512][n]
__global__ void k_build_w01T(const float* __restrict__ Wg, _Float16* __restrict__ out)
{
    int idx = blockIdx.x * 256 + threadIdx.x;          // 512*1024
    if (idx >= 512 * 1024) return;
    int n = idx >> 10, k = idx & 1023;
    float v = (k < 512) ? Wg[(size_t)k * 512 + n]
                        : Wg[(size_t)512 * 512 + (size_t)(k - 512) * 512 + n];
    out[(size_t)n * 1024 + k] = (_Float16)v;
}

// ---------------------------------------------------------------------------
// Adjacency softmax: A = softmax(logits), logits = NEG except e at (r,c)
// ---------------------------------------------------------------------------
__global__ void k_build_A(const float* __restrict__ e, const int* __restrict__ rows,
                          const int* __restrict__ cols, int nnz, float* __restrict__ Amat)
{
    __shared__ float L[JJ * JJ];
    int t = threadIdx.x;
    for (int i = t; i < JJ * JJ; i += 32) L[i] = -9e15f;
    __syncthreads();
    for (int i = t; i < nnz; i += 32) L[rows[i] * JJ + cols[i]] = e[i];
    __syncthreads();
    if (t < JJ) {
        float mx = -3.0e38f;
        for (int c = 0; c < JJ; ++c) mx = fmaxf(mx, L[t * JJ + c]);
        float s = 0.0f;
        for (int c = 0; c < JJ; ++c) s += expf(L[t * JJ + c] - mx);
        float inv = 1.0f / s;
        for (int c = 0; c < JJ; ++c) Amat[t * JJ + c] = expf(L[t * JJ + c] - mx) * inv;
    }
}

// ---------------------------------------------------------------------------
// Attention: one wave per (b,h). qkv f16 rows of 1536 = [Q|K|V], per-head 64.
// ---------------------------------------------------------------------------
__global__ __launch_bounds__(32)
void k_attention(const _Float16* __restrict__ qkv, _Float16* __restrict__ attn)
{
    const int bh = blockIdx.x;
    const int b = bh >> 3, h = bh & 7;
    const int lane = threadIdx.x;
    __shared__ float Ks[JJ][DK];
    __shared__ float Vs[JJ][DK];

    const _Float16* base = qkv + (size_t)(b * JJ) * 1536 + h * DK;
    for (int idx = lane; idx < JJ * DK; idx += 32) {
        int j = idx >> 6, ee = idx & 63;
        Ks[j][ee] = (float)base[(size_t)j * 1536 + 512 + ee];
        Vs[j][ee] = (float)base[(size_t)j * 1536 + 1024 + ee];
    }
    __syncthreads();

    if (lane < JJ) {
        float q[DK];
        const _Float16* qp = base + (size_t)lane * 1536;
        #pragma unroll
        for (int ee = 0; ee < DK; ++ee) q[ee] = (float)qp[ee];

        float sc[JJ]; float mx = -3.0e38f;
        for (int k = 0; k < JJ; ++k) {
            float s = 0.0f;
            #pragma unroll
            for (int ee = 0; ee < DK; ++ee) s += q[ee] * Ks[k][ee];
            s *= 0.125f;                       // 1/sqrt(64)
            sc[k] = s; mx = fmaxf(mx, s);
        }
        float sum = 0.0f;
        for (int k = 0; k < JJ; ++k) { sc[k] = expf(sc[k] - mx); sum += sc[k]; }
        float inv = 1.0f / sum;

        _Float16* op = attn + (size_t)(b * JJ + lane) * 512 + h * DK;
        #pragma unroll
        for (int ee = 0; ee < DK; ++ee) {
            float o = 0.0f;
            for (int k = 0; k < JJ; ++k) o += sc[k] * Vs[k][ee];
            op[ee] = (_Float16)(o * inv);
        }
    }
}

// ---------------------------------------------------------------------------
// LayerNorm: out = LN(resid + raw) * g + b.  One wave per 512-wide row.
// ---------------------------------------------------------------------------
__global__ __launch_bounds__(256)
void k_layernorm(const float* __restrict__ resid, const float* __restrict__ raw,
                 const float* __restrict__ g, const float* __restrict__ bta,
                 float* __restrict__ out, int nrows)
{
    int row = blockIdx.x * 8 + (threadIdx.x >> 5);
    int lane = threadIdx.x & 31;
    if (row >= nrows) return;
    const float* pr = resid + (size_t)row * 512;
    const float* pa = raw   + (size_t)row * 512;

    float v[16]; float s = 0.0f;
    #pragma unroll
    for (int i = 0; i < 16; ++i) { v[i] = pr[lane + i * 32] + pa[lane + i * 32]; s += v[i]; }
    #pragma unroll
    for (int off = 16; off > 0; off >>= 1) s += __shfl_xor(s, off, 32);
    float mu = s * (1.0f / 512.0f);

    float sv = 0.0f;
    #pragma unroll
    for (int i = 0; i < 16; ++i) { float d = v[i] - mu; sv += d * d; }
    #pragma unroll
    for (int off = 16; off > 0; off >>= 1) sv += __shfl_xor(sv, off, 32);
    float rstd = rsqrtf(sv * (1.0f / 512.0f) + 1e-5f);

    float* po = out + (size_t)row * 512;
    #pragma unroll
    for (int i = 0; i < 16; ++i) {
        int c = lane + i * 32;
        po[c] = (v[i] - mu) * rstd * g[c] + bta[c];
    }
}

// ---------------------------------------------------------------------------
// GCN pre-mix: y0[b,j,:] = A[j,j]*x[b,j,:]; y1[b,j,:] = sum_{k!=j} A[j,k]*x[b,k,:]
// Output y01 f16 [M x 1024] = [y0 | y1].  One block per batch b.
// ---------------------------------------------------------------------------
__global__ __launch_bounds__(256)
void k_gcn_mix(const float* __restrict__ X, const float* __restrict__ Amat,
               _Float16* __restrict__ y01)
{
    const int b = blockIdx.x;
    const int t = threadIdx.x;
    __shared__ float xs[JJ][DD];
    __shared__ float As[JJ * JJ];

    for (int i = t; i < JJ * JJ; i += 256) As[i] = Amat[i];
    const float* xb = X + (size_t)b * JJ * DD;
    for (int i = t; i < JJ * DD; i += 256) ((float*)xs)[i] = xb[i];
    __syncthreads();

    #pragma unroll
    for (int cc = 0; cc < 2; ++cc) {
        int col = t + cc * 256;
        for (int j = 0; j < JJ; ++j) {
            float y0 = As[j * JJ + j] * xs[j][col];
            float y1 = 0.0f;
            #pragma unroll
            for (int k = 0; k < JJ; ++k)
                if (k != j) y1 += As[j * JJ + k] * xs[k][col];
            _Float16* yp = y01 + (size_t)(b * JJ + j) * 1024;
            yp[col]       = (_Float16)y0;
            yp[512 + col] = (_Float16)y1;
        }
    }
}

// ---------------------------------------------------------------------------
extern "C" void kernel_launch(void* const* d_in, const int* in_sizes, int n_in,
                              void* d_out, int out_size, void* d_ws, size_t ws_size,
                              hipStream_t stream)
{
    const float* src  = (const float*)d_in[0];
    const float* Wq   = (const float*)d_in[1];
    const float* bq   = (const float*)d_in[2];
    const float* Wk   = (const float*)d_in[3];
    const float* bk   = (const float*)d_in[4];
    const float* Wv   = (const float*)d_in[5];
    const float* bv   = (const float*)d_in[6];
    const float* Wo   = (const float*)d_in[7];
    const float* bo   = (const float*)d_in[8];
    const float* l1g  = (const float*)d_in[9];
    const float* l1b  = (const float*)d_in[10];
    const float* Wg   = (const float*)d_in[11];
    const float* eg   = (const float*)d_in[12];
    const float* bg   = (const float*)d_in[13];
    const float* l2g  = (const float*)d_in[14];
    const float* l2b  = (const float*)d_in[15];
    const int*   mrow = (const int*)d_in[16];
    const int*   mcol = (const int*)d_in[17];
    const int nnz = in_sizes[12];

    // workspace carve
    char* w = (char*)d_ws;
    size_t off = 0;
    auto carve = [&](size_t bytes) -> char* {
        char* p = w + off;
        off += (bytes + 255) & ~(size_t)255;
        return p;
    };
    float*    O1     = (float*)   carve((size_t)MM * 512 * 4); // proj_raw, later gcn_raw
    float*    Xb     = (float*)   carve((size_t)MM * 512 * 4); // x after LN1
    _Float16* S16    = (_Float16*)carve((size_t)MM * 512 * 2); // src f16, later attn f16
    _Float16* QKV16  = (_Float16*)carve((size_t)MM * 1536 * 2);// qkv f16, later y01 f16
    _Float16* WqkvT  = (_Float16*)carve((size_t)1536 * 512 * 2);
    float*    bqkv   = (float*)   carve(1536 * 4);
    _Float16* WoT    = (_Float16*)carve((size_t)512 * 512 * 2);
    _Float16* W01T   = (_Float16*)carve((size_t)512 * 1024 * 2);
    float*    Amat   = (float*)   carve(JJ * JJ * 4);
    _Float16* Y01    = QKV16;     // reuse (qkv dead after attention)
    _Float16* ATT16  = S16;       // reuse (src f16 dead after qkv gemm)

    // --- prep ---
    k_cvt_src<<<(MM * 512) / 256, 256, 0, stream>>>(src, S16, MM * 512);
    k_build_wqkvT<<<(1536 * 512) / 256, 256, 0, stream>>>(Wq, Wk, Wv, WqkvT);
    k_build_bqkv<<<6, 256, 0, stream>>>(bq, bk, bv, bqkv);
    k_build_woT<<<(512 * 512) / 256, 256, 0, stream>>>(Wo, WoT);
    k_build_w01T<<<(512 * 1024) / 256, 256, 0, stream>>>(Wg, W01T);
    k_build_A<<<1, 32, 0, stream>>>(eg, mrow, mcol, nnz, Amat);

    // --- QKV projection: [M x 512] x [512 x 1536] -> f16 [M x 1536] ---
    gemm_wmma_f16<1><<<dim3(MM / 128, 1536 / 128), 256, 0, stream>>>(
        S16, 512, WqkvT, 512, bqkv, (void*)QKV16, 1536, 512);

    // --- attention per (b,h) ---
    k_attention<<<BB * HH, 32, 0, stream>>>(QKV16, ATT16);

    // --- output projection: [M x 512] x [512 x 512] + bo -> f32 O1 ---
    gemm_wmma_f16<0><<<dim3(MM / 128, 512 / 128), 256, 0, stream>>>(
        ATT16, 512, WoT, 512, bo, (void*)O1, 512, 512);

    // --- LN1: x = LN(src + O1) ---
    k_layernorm<<<MM / 8, 256, 0, stream>>>(src, O1, l1g, l1b, Xb, MM);

    // --- GCN mix -> y01 f16 [M x 1024] ---
    k_gcn_mix<<<BB, 256, 0, stream>>>(Xb, Amat, Y01);

    // --- GCN GEMM: [M x 1024] x [1024 x 512] + b_gcn -> f32 O1 ---
    gemm_wmma_f16<0><<<dim3(MM / 128, 512 / 128), 256, 0, stream>>>(
        Y01, 1024, W01T, 1024, bg, (void*)O1, 512, 1024);

    // --- LN2: out = LN(x + O1) ---
    k_layernorm<<<MM / 8, 256, 0, stream>>>(Xb, O1, l2g, l2b, (float*)d_out, MM);
}
